// TokenSeq2SeqMorphSeg_58016418234655
// MI455X (gfx1250) — matrix operand order repaired and checked by reading, not compiled, not run on hardware
//
#include <hip/hip_runtime.h>

// ---------------------------------------------------------------------------
// Problem constants (match the reference)
// ---------------------------------------------------------------------------
static constexpr int   Vv  = 50257;   // vocab
static constexpr int   Ee  = 768;     // embed dim
static constexpr int   HD  = 1024;    // hidden dim
static constexpr int   Tt  = 4096;    // encoder steps
static constexpr int   ML  = 1024;    // decoder steps (max_len)
static constexpr int   G3  = 3 * HD;  // 3072 gate rows

// ---------------------------------------------------------------------------
// Workspace layout (all offsets 256B aligned by construction)
// ---------------------------------------------------------------------------
static constexpr size_t o_bar   = 0;                                        // 4 barrier slots
static constexpr size_t o_xe    = o_bar   + 1024;                           // enc emb bf16 [T,E]
static constexpr size_t o_xd    = o_xe    + (size_t)Tt * Ee * 2;            // dec emb bf16 [ML,E]
static constexpr size_t o_gi    = o_xd    + (size_t)ML * Ee * 2;            // gi fp32 [T,3H] (reused)
static constexpr size_t o_s0    = o_gi    + (size_t)Tt * G3 * 4;            // seq bf16 [T,H]
static constexpr size_t o_s1    = o_s0    + (size_t)Tt * HD * 2;            // seq bf16 [T,H]
static constexpr size_t o_hb    = o_s1    + (size_t)Tt * HD * 2;            // h double buffer fp32 [2,H]
static constexpr size_t o_hT    = o_hb    + 2 * HD * 4;                     // 4 final-h slots fp32
static constexpr size_t o_wih0e = o_hT    + 4 * HD * 4;                     // bf16 weights ...
static constexpr size_t o_whh0e = o_wih0e + (size_t)G3 * Ee * 2;
static constexpr size_t o_wih1e = o_whh0e + (size_t)G3 * HD * 2;
static constexpr size_t o_whh1e = o_wih1e + (size_t)G3 * HD * 2;
static constexpr size_t o_wih0d = o_whh1e + (size_t)G3 * HD * 2;
static constexpr size_t o_whh0d = o_wih0d + (size_t)G3 * Ee * 2;
static constexpr size_t o_wih1d = o_whh0d + (size_t)G3 * HD * 2;
static constexpr size_t o_whh1d = o_wih1d + (size_t)G3 * HD * 2;
static constexpr size_t o_outw  = o_whh1d + (size_t)G3 * HD * 2;            // bf16 out_w [V,H]

// ---------------------------------------------------------------------------
// Helpers
// ---------------------------------------------------------------------------
__device__ __forceinline__ unsigned short f2bf(float f) {
  union { float f; unsigned int u; } c; c.f = f;
  unsigned int u = c.u;
  unsigned int r = ((u >> 16) & 1u) + 0x7FFFu;       // round-to-nearest-even
  return (unsigned short)((u + r) >> 16);
}
__device__ __forceinline__ float bflo(unsigned int u) {
  union { unsigned int x; float f; } c; c.x = u << 16; return c.f;
}
__device__ __forceinline__ float bfhi(unsigned int u) {
  union { unsigned int x; float f; } c; c.x = u & 0xffff0000u; return c.f;
}

// ---------------------------------------------------------------------------
// fp32 -> bf16 bulk conversion
// ---------------------------------------------------------------------------
__global__ void f32_to_bf16_k(const float* __restrict__ src,
                              unsigned short* __restrict__ dst, size_t n) {
  size_t i = (size_t)blockIdx.x * blockDim.x + threadIdx.x;
  if (i < n) dst[i] = f2bf(src[i]);
}

// ---------------------------------------------------------------------------
// Embedding gathers (fp32 table -> bf16 rows)
// ---------------------------------------------------------------------------
__global__ void gather_embed_k(const int* __restrict__ ids,
                               const float* __restrict__ emb,
                               unsigned short* __restrict__ dst,
                               int rows, int cols) {
  size_t i = (size_t)blockIdx.x * blockDim.x + threadIdx.x;
  size_t n = (size_t)rows * cols;
  if (i >= n) return;
  int r = (int)(i / cols), c = (int)(i % cols);
  dst[i] = f2bf(emb[(size_t)ids[r] * cols + c]);
}

// decoder inputs: [sos, target[0..ML-2]]
__global__ void gather_embed_dec_k(const int* __restrict__ sos,
                                   const int* __restrict__ target,
                                   const float* __restrict__ emb,
                                   unsigned short* __restrict__ dst,
                                   int rows, int cols) {
  size_t i = (size_t)blockIdx.x * blockDim.x + threadIdx.x;
  size_t n = (size_t)rows * cols;
  if (i >= n) return;
  int r = (int)(i / cols), c = (int)(i % cols);
  int id = (r == 0) ? sos[0] : target[r - 1];
  dst[i] = f2bf(emb[(size_t)id * cols + c]);
}

// ---------------------------------------------------------------------------
// WMMA bf16 GEMM:  C[M,N] = A[M,K] * W[N,K]^T + bias[N]     (fp32 out)
// Block = 1 wave; wave computes a 32(M) x 64(N) tile: 2 A fragments are
// reused across 4 B fragments (8 x v_wmma_f32_16x16x32_bf16 per K-step).
// Fragments are loaded as explicit b128 pairs (all 16B aligned).
// B rows for n >= N are clamped to row 0: garbage only lands in output
// columns that the (guarded) store never writes.
// ---------------------------------------------------------------------------
typedef __bf16 v16bf __attribute__((ext_vector_type(16)));
typedef float  v8f   __attribute__((ext_vector_type(8)));

__device__ __forceinline__ v16bf ld_frag_a(const unsigned short* __restrict__ p,
                                           int kb, int half) {
  union { uint4 q[2]; v16bf v; } f;
  const unsigned short* b = p + kb + (half << 3);
  f.q[0] = *(const uint4*)(b);        // K = kb+half*8   .. +7   (pairs v0..v3)
  f.q[1] = *(const uint4*)(b + 16);   // K = kb+16+half*8 .. +7  (pairs v4..v7)
  return f.v;
}
__device__ __forceinline__ v16bf ld_frag_b(const unsigned short* __restrict__ p,
                                           int kb, int half) {
  union { uint4 q[2]; v16bf v; } f;
  const unsigned short* b = p + kb + (half << 3);
  f.q[0] = *(const uint4*)(b);        // K = kb+half*8   .. +7
  f.q[1] = *(const uint4*)(b + 16);   // K = kb+16+half*8 .. +7
  return f.v;
}

__global__ __launch_bounds__(32) void wmma_gemm_bf16_k(
    const unsigned short* __restrict__ A, int lda,
    const unsigned short* __restrict__ W, int ldw,
    const float* __restrict__ bias,
    float* __restrict__ C,
    int M, int N, int K) {
  const int lane = threadIdx.x & 31;
  const int half = lane >> 4;
  const int idx  = lane & 15;
  const int m0 = blockIdx.y << 5;   // 32 rows per block
  const int n0 = blockIdx.x << 6;   // 64 cols per block

  v8f acc[2][4];
#pragma unroll
  for (int ms = 0; ms < 2; ++ms)
#pragma unroll
    for (int s = 0; s < 4; ++s)
#pragma unroll
      for (int d = 0; d < 8; ++d) acc[ms][s][d] = 0.0f;

  const unsigned short* Arow0 = A + (size_t)(m0 + idx) * lda;
  const unsigned short* Arow1 = A + (size_t)(m0 + 16 + idx) * lda;
  const unsigned short* Wrow[4];
#pragma unroll
  for (int s = 0; s < 4; ++s) {
    int n = n0 + (s << 4) + idx;
    Wrow[s] = W + (size_t)((n < N) ? n : 0) * ldw;   // clamp, load unconditionally
  }

  for (int kb = 0; kb < K; kb += 32) {
    v16bf a0 = ld_frag_a(Arow0, kb, half);
    v16bf a1 = ld_frag_a(Arow1, kb, half);
#pragma unroll
    for (int s = 0; s < 4; ++s) {
      v16bf b = ld_frag_b(Wrow[s], kb, half);
      acc[0][s] = __builtin_amdgcn_wmma_f32_16x16x32_bf16(
          false, a0, false, b, (short)0, acc[0][s], false, false);
      acc[1][s] = __builtin_amdgcn_wmma_f32_16x16x32_bf16(
          false, a1, false, b, (short)0, acc[1][s], false, false);
    }
  }

#pragma unroll
  for (int ms = 0; ms < 2; ++ms)
#pragma unroll
    for (int s = 0; s < 4; ++s) {
      int n = n0 + (s << 4) + idx;
      if (n >= N) continue;
      float bv = bias[n];
#pragma unroll
      for (int d = 0; d < 8; ++d) {
        int m = m0 + (ms << 4) + d + (half << 3);
        C[(size_t)m * N + n] = acc[ms][s][d] + bv;
      }
    }
}

// ---------------------------------------------------------------------------
// Persistent GRU recurrence.
//   8 workgroups x 128 threads = 1024 threads; thread j owns hidden unit j
//   (rows j, H+j, 2H+j of w_hh -> no cross-thread gate exchange).
//   h double-buffered in global, mirrored to LDS each step; one device-scope
//   barrier per step. w_hh is bf16 (L2 resident), gate math fp32.
// ---------------------------------------------------------------------------
#define SCAN_WG   8
#define SCAN_TPB  128

__device__ __forceinline__ void grid_barrier(unsigned int* bar,
                                             unsigned int nwg,
                                             unsigned int target) {
  __threadfence();
  __syncthreads();
  if (threadIdx.x == 0) {
    unsigned int old = __hip_atomic_fetch_add(&bar[0], 1u, __ATOMIC_ACQ_REL,
                                              __HIP_MEMORY_SCOPE_AGENT);
    if (old == nwg - 1u) {
      __hip_atomic_store(&bar[0], 0u, __ATOMIC_RELAXED, __HIP_MEMORY_SCOPE_AGENT);
      __hip_atomic_fetch_add(&bar[1], 1u, __ATOMIC_ACQ_REL, __HIP_MEMORY_SCOPE_AGENT);
    } else {
      while (__hip_atomic_load(&bar[1], __ATOMIC_ACQUIRE,
                               __HIP_MEMORY_SCOPE_AGENT) < target) {
        __builtin_amdgcn_s_sleep(1);
      }
    }
  }
  __syncthreads();
}

__device__ __forceinline__ float dot_row_bf16(const uint4* __restrict__ Wr,
                                              const float* __restrict__ sh) {
  float a = 0.f;
#pragma unroll 4
  for (int q = 0; q < HD / 8; ++q) {
    uint4 w = Wr[q];
    const float* hs = sh + (q << 3);
    a += bflo(w.x) * hs[0] + bfhi(w.x) * hs[1]
       + bflo(w.y) * hs[2] + bfhi(w.y) * hs[3]
       + bflo(w.z) * hs[4] + bfhi(w.z) * hs[5]
       + bflo(w.w) * hs[6] + bfhi(w.w) * hs[7];
  }
  return a;
}

__global__ __launch_bounds__(SCAN_TPB) void gru_scan_k(
    const float* __restrict__ gi,            // [T, 3H] fp32 (x @ w_ih^T + b_ih)
    const unsigned short* __restrict__ whh,  // [3H, H] bf16
    const float* __restrict__ bhh,           // [3H]
    const float* __restrict__ h0,            // [H]
    int T,
    unsigned short* __restrict__ seq,        // [T, H] bf16 out
    float* __restrict__ hT,                  // [H] final state
    float* __restrict__ hbuf,                // [2, H] double buffer
    unsigned int* __restrict__ bar) {
  __shared__ float sh[HD];
  const int j = blockIdx.x * blockDim.x + threadIdx.x;   // 0..1023

  hbuf[j] = h0[j];
  const float b0 = bhh[j], b1 = bhh[HD + j], b2 = bhh[2 * HD + j];
  const uint4* W0 = (const uint4*)(whh + (size_t)j * HD);
  const uint4* W1 = (const uint4*)(whh + (size_t)(HD + j) * HD);
  const uint4* W2 = (const uint4*)(whh + (size_t)(2 * HD + j) * HD);

  unsigned int step = 1;
  grid_barrier(bar, gridDim.x, step++);   // h0 visible everywhere

  float hnew = 0.f;
  for (int t = 0; t < T; ++t) {
    const float* hcur = hbuf + (t & 1) * HD;
    for (int i = threadIdx.x; i < HD; i += blockDim.x) sh[i] = hcur[i];
    __syncthreads();

    float a0 = b0 + dot_row_bf16(W0, sh);
    float a1 = b1 + dot_row_bf16(W1, sh);
    float a2 = b2 + dot_row_bf16(W2, sh);

    const float* git = gi + (size_t)t * G3;
    float r = 1.f / (1.f + __expf(-(git[j] + a0)));
    float z = 1.f / (1.f + __expf(-(git[HD + j] + a1)));
    float n = tanhf(git[2 * HD + j] + r * a2);
    float hold = sh[j];
    hnew = (1.f - z) * n + z * hold;

    hbuf[((t + 1) & 1) * HD + j] = hnew;
    seq[(size_t)t * HD + j] = f2bf(hnew);
    grid_barrier(bar, gridDim.x, step++);
  }
  hT[j] = hnew;
}

// ---------------------------------------------------------------------------
// Host driver
// ---------------------------------------------------------------------------
extern "C" void kernel_launch(void* const* d_in, const int* in_sizes, int n_in,
                              void* d_out, int out_size, void* d_ws, size_t ws_size,
                              hipStream_t stream) {
  (void)in_sizes; (void)n_in; (void)out_size; (void)ws_size;

  const int*   ids_enc = (const int*)d_in[0];
  const float* state   = (const float*)d_in[1];
  const int*   sos     = (const int*)d_in[2];
  const int*   target  = (const int*)d_in[5];
  const float* emb     = (const float*)d_in[6];
  const float* out_w   = (const float*)d_in[7];
  const float* out_b   = (const float*)d_in[8];
  const float* ewih0 = (const float*)d_in[9],  *ewhh0 = (const float*)d_in[10];
  const float* ebih0 = (const float*)d_in[11], *ebhh0 = (const float*)d_in[12];
  const float* ewih1 = (const float*)d_in[13], *ewhh1 = (const float*)d_in[14];
  const float* ebih1 = (const float*)d_in[15], *ebhh1 = (const float*)d_in[16];
  const float* dwih0 = (const float*)d_in[17], *dwhh0 = (const float*)d_in[18];
  const float* dbih0 = (const float*)d_in[19], *dbhh0 = (const float*)d_in[20];
  const float* dwih1 = (const float*)d_in[21], *dwhh1 = (const float*)d_in[22];
  const float* dbih1 = (const float*)d_in[23], *dbhh1 = (const float*)d_in[24];

  char* ws = (char*)d_ws;
  unsigned int*   bar0 = (unsigned int*)(ws + o_bar + 0);
  unsigned int*   bar1 = (unsigned int*)(ws + o_bar + 256);
  unsigned int*   bar2 = (unsigned int*)(ws + o_bar + 512);
  unsigned int*   bar3 = (unsigned int*)(ws + o_bar + 768);
  unsigned short* Xe   = (unsigned short*)(ws + o_xe);
  unsigned short* Xd   = (unsigned short*)(ws + o_xd);
  float*          gi   = (float*)(ws + o_gi);
  unsigned short* s0   = (unsigned short*)(ws + o_s0);
  unsigned short* s1   = (unsigned short*)(ws + o_s1);
  float*          hbuf = (float*)(ws + o_hb);
  float*          hT0  = (float*)(ws + o_hT);
  float*          hT1  = hT0 + HD;
  float*          hT2  = hT0 + 2 * HD;
  float*          hT3  = hT0 + 3 * HD;
  unsigned short* Wih0e = (unsigned short*)(ws + o_wih0e);
  unsigned short* Whh0e = (unsigned short*)(ws + o_whh0e);
  unsigned short* Wih1e = (unsigned short*)(ws + o_wih1e);
  unsigned short* Whh1e = (unsigned short*)(ws + o_whh1e);
  unsigned short* Wih0d = (unsigned short*)(ws + o_wih0d);
  unsigned short* Whh0d = (unsigned short*)(ws + o_whh0d);
  unsigned short* Wih1d = (unsigned short*)(ws + o_wih1d);
  unsigned short* Whh1d = (unsigned short*)(ws + o_whh1d);
  unsigned short* OutW  = (unsigned short*)(ws + o_outw);

  hipMemsetAsync(ws + o_bar, 0, 1024, stream);

  auto conv = [&](const float* s, unsigned short* dptr, size_t n) {
    unsigned int blocks = (unsigned int)((n + 255) / 256);
    f32_to_bf16_k<<<blocks, 256, 0, stream>>>(s, dptr, n);
  };
  conv(ewih0, Wih0e, (size_t)G3 * Ee);
  conv(ewhh0, Whh0e, (size_t)G3 * HD);
  conv(ewih1, Wih1e, (size_t)G3 * HD);
  conv(ewhh1, Whh1e, (size_t)G3 * HD);
  conv(dwih0, Wih0d, (size_t)G3 * Ee);
  conv(dwhh0, Whh0d, (size_t)G3 * HD);
  conv(dwih1, Wih1d, (size_t)G3 * HD);
  conv(dwhh1, Whh1d, (size_t)G3 * HD);
  conv(out_w, OutW,  (size_t)Vv * HD);

  {
    size_t n = (size_t)Tt * Ee;
    gather_embed_k<<<(unsigned)((n + 255) / 256), 256, 0, stream>>>(
        ids_enc, emb, Xe, Tt, Ee);
    size_t nd = (size_t)ML * Ee;
    gather_embed_dec_k<<<(unsigned)((nd + 255) / 256), 256, 0, stream>>>(
        sos, target, emb, Xd, ML, Ee);
  }

  // ---- encoder layer 0
  wmma_gemm_bf16_k<<<dim3(G3 / 64, Tt / 32), 32, 0, stream>>>(
      Xe, Ee, Wih0e, Ee, ebih0, gi, Tt, G3, Ee);
  gru_scan_k<<<SCAN_WG, SCAN_TPB, 0, stream>>>(
      gi, Whh0e, ebhh0, state + 0, Tt, s0, hT0, hbuf, bar0);

  // ---- encoder layer 1
  wmma_gemm_bf16_k<<<dim3(G3 / 64, Tt / 32), 32, 0, stream>>>(
      s0, HD, Wih1e, HD, ebih1, gi, Tt, G3, HD);
  gru_scan_k<<<SCAN_WG, SCAN_TPB, 0, stream>>>(
      gi, Whh1e, ebhh1, state + HD, Tt, s1, hT1, hbuf, bar1);

  // ---- decoder layer 0 (h0 = encoder layer-0 final state)
  wmma_gemm_bf16_k<<<dim3(G3 / 64, ML / 32), 32, 0, stream>>>(
      Xd, Ee, Wih0d, Ee, dbih0, gi, ML, G3, Ee);
  gru_scan_k<<<SCAN_WG, SCAN_TPB, 0, stream>>>(
      gi, Whh0d, dbhh0, hT0, ML, s0, hT2, hbuf, bar2);

  // ---- decoder layer 1 (h0 = encoder layer-1 final state)
  wmma_gemm_bf16_k<<<dim3(G3 / 64, ML / 32), 32, 0, stream>>>(
      s0, HD, Wih1d, HD, dbih1, gi, ML, G3, HD);
  gru_scan_k<<<SCAN_WG, SCAN_TPB, 0, stream>>>(
      gi, Whh1d, dbhh1, hT1, ML, s1, hT3, hbuf, bar3);

  // ---- output projection: [1024,1024] x [1024,50257] + out_b -> d_out fp32
  wmma_gemm_bf16_k<<<dim3((Vv + 63) / 64, ML / 32), 32, 0, stream>>>(
      s1, HD, OutW, HD, out_b, (float*)d_out, ML, Vv, HD);
}
